// STGCNClassifier_38989713113523
// MI455X (gfx1250) — matrix operand order repaired and checked by reading, not compiled, not run on hardware
//
#include <hip/hip_runtime.h>
#include <hip/hip_bf16.h>

typedef __bf16 bf16_t;
typedef __attribute__((ext_vector_type(16))) __bf16 v16bf;
typedef __attribute__((ext_vector_type(8)))  __bf16 v8bf;
typedef __attribute__((ext_vector_type(8)))  float  v8f;
typedef unsigned int uint32;

// ---------------------------------------------------------------------------
// WMMA fragment helpers (CDNA5 16x16x32 bf16, wave32).
// load_A_frag : operand element (m,k) from row-major [m][k] storage (contig k)
// load_Bt_frag: operand element (k,n) from N-major  [n][k] storage (contig k)
// Both give 16B-contiguous per-lane runs -> b128 loads.
// C/D f32 frag: lane L, VGPR v -> (M = 8*(L>>4)+v, N = L&15); contiguous when
// stored [n][m] (v8bf packed stores).
// ---------------------------------------------------------------------------
__device__ __forceinline__ v16bf load_A_frag(const bf16_t* base, int ld, int k0) {
  int lane = threadIdx.x & 31;
  int m = lane & 15, h = lane >> 4;
  v16bf a;
#pragma unroll
  for (int v = 0; v < 8; ++v) {
    int k = k0 + (h << 4) + (v << 1);
    a[2 * v]     = base[m * ld + k];
    a[2 * v + 1] = base[m * ld + k + 1];
  }
  return a;
}

__device__ __forceinline__ v16bf load_Bt_frag(const bf16_t* base, int ld, int k0, int n0) {
  int lane = threadIdx.x & 31;
  int n = n0 + (lane & 15), h = lane >> 4;
  v16bf b;
#pragma unroll
  for (int v = 0; v < 8; ++v) {
    int k = k0 + (h << 4) + (v << 1);
    b[2 * v]     = base[n * ld + k];
    b[2 * v + 1] = base[n * ld + k + 1];
  }
  return b;
}

__device__ __forceinline__ v8f wmma_bf16(v16bf a, v16bf b, v8f c) {
  return __builtin_amdgcn_wmma_f32_16x16x32_bf16(false, a, false, b, (short)0, c,
                                                 false, false);
}

__device__ __forceinline__ float sigmoidf_(float x) {
  return 1.0f / (1.0f + __expf(-x));
}

// ---------------------------------------------------------------------------
// Prep: dense normalized Laplacian (33x33 -> 64x64 bf16 row-major [dst][src]),
// temporal-conv weights transposed to [gate][c_out=64][K=192],
// cheb Wcat transposed to [d=64][K=192], fc1 weights transposed to [256][2112].
// ---------------------------------------------------------------------------
__global__ void prep_kernel(const int* __restrict__ edge,
                            const float* __restrict__ s1tc2w,
                            const float* __restrict__ s2tc1w,
                            const float* __restrict__ s2tc2w,
                            const float* __restrict__ cheb1w,
                            const float* __restrict__ cheb2w,
                            const float* __restrict__ fc1w,
                            bf16_t* __restrict__ Lb,
                            bf16_t* __restrict__ wtcT,
                            bf16_t* __restrict__ wcatT1,
                            bf16_t* __restrict__ wcatT2,
                            bf16_t* __restrict__ fc1wT) {
  int tid = blockIdx.x * blockDim.x + threadIdx.x;
  int nth = gridDim.x * blockDim.x;

  if (tid == 0) {
    float deg[33];
    float Lf[33 * 33];
    for (int i = 0; i < 33; ++i) deg[i] = 0.f;
    for (int i = 0; i < 33 * 33; ++i) Lf[i] = 0.f;
    for (int e = 0; e < 64; ++e) deg[edge[e]] += 1.f;
    float dinv[33];
    for (int i = 0; i < 33; ++i) dinv[i] = (deg[i] > 0.f) ? rsqrtf(deg[i]) : 0.f;
    for (int e = 0; e < 64; ++e) {
      int s = edge[e], t = edge[64 + e];
      Lf[t * 33 + s] += -dinv[s] * dinv[t];
    }
    for (int i = 0; i < 64 * 64; ++i) {
      int r = i >> 6, c = i & 63;
      Lb[i] = (r < 33 && c < 33) ? (bf16_t)Lf[r * 33 + c] : (bf16_t)0.0f;
    }
  }

  // wtcT[tensor][g][o][k], k = kk*64+ci ; src shape (3,O,Ci,1,ks)
  const float* tsrc[3] = {s1tc2w, s2tc1w, s2tc2w};
  for (int i = tid; i < 3 * 36864; i += nth) {
    int tix = i / 36864, j = i % 36864;
    int g = j / 12288, o = (j / 192) % 64, k = j % 192;
    int kk = k >> 6, ci = k & 63;
    wtcT[i] = (bf16_t)tsrc[tix][((g * 64 + o) * 64 + ci) * 3 + kk];
  }
  // cheb W (3,64,64) row-major is [K=192][d] -> transpose to [d][192]
  for (int i = tid; i < 64 * 192; i += nth) {
    int d = i / 192, k = i % 192;
    wcatT1[i] = (bf16_t)cheb1w[k * 64 + d];
    wcatT2[i] = (bf16_t)cheb2w[k * 64 + d];
  }
  // fc1 (2112,256) -> [256][2112]
  for (int i = tid; i < 256 * 2112; i += nth) {
    int o = i / 2112, k = i % 2112;
    fc1wT[i] = (bf16_t)fc1w[k * 256 + o];
  }
}

// ---------------------------------------------------------------------------
// First temporal conv (Cin=3 -> VALU), gated: relu(P*sig(Q)+R)
// ---------------------------------------------------------------------------
__global__ void tconv3_kernel(const float* __restrict__ x,
                              const float* __restrict__ w,
                              const float* __restrict__ bias,
                              bf16_t* __restrict__ out, int B, int Tin) {
  const int N = 33;
  int c = threadIdx.x;
  int n = blockIdx.y;
  int Tout = Tin - 2;
  int bt = blockIdx.x;
  int b = bt / Tout, t = bt % Tout;
  float xv[3][3];
#pragma unroll
  for (int kk = 0; kk < 3; ++kk)
#pragma unroll
    for (int ci = 0; ci < 3; ++ci)
      xv[kk][ci] = x[(((size_t)b * Tin + (t + kk)) * N + n) * 3 + ci];
  float acc[3];
#pragma unroll
  for (int g = 0; g < 3; ++g) {
    float a = bias[g * 64 + c];
#pragma unroll
    for (int ci = 0; ci < 3; ++ci)
#pragma unroll
      for (int kk = 0; kk < 3; ++kk)
        a += w[((g * 64 + c) * 3 + ci) * 3 + kk] * xv[kk][ci];
    acc[g] = a;
  }
  float h = fmaxf(acc[0] * sigmoidf_(acc[1]) + acc[2], 0.f);
  out[(((size_t)b * Tout + t) * N + n) * 64 + c] = (bf16_t)h;
}

// ---------------------------------------------------------------------------
// Gated temporal conv, Cin=64, WMMA, transposed orientation:
// Y^T[c][t'] = sum_k Wg^T[c][k] * unfold[t'][k].
// A = wT[gate][c][192] direct from global; B = LDS input slab (shifted rows).
// Wave w owns output-channel tile [16w,16w+16). Merged 16B output stores.
// ---------------------------------------------------------------------------
__global__ __launch_bounds__(128) void tconv_gemm_kernel(
    const bf16_t* __restrict__ in, const bf16_t* __restrict__ wT,
    const float* __restrict__ bias, bf16_t* __restrict__ out, int B, int Tin) {
  const int N = 33;
  int Tout = Tin - 2;
  int t0 = blockIdx.x * 16;
  int n = blockIdx.y;
  int b = blockIdx.z;

  __shared__ __align__(16) bf16_t sA[18 * 64];

  int tid = threadIdx.x;
  const uint32* in32 = (const uint32*)in;
  uint32* sA32 = (uint32*)sA;
  for (int i = tid; i < 18 * 32; i += 128) {
    int r = i >> 5, cc = i & 31;
    int t = t0 + r;
    sA32[i] = (t < Tin) ? in32[(((size_t)b * Tin + t) * N + n) * 32 + cc] : 0u;
  }
  __syncthreads();

  int w = tid >> 5, lane = tid & 31;
  v8f accP = {0, 0, 0, 0, 0, 0, 0, 0};
  v8f accQ = {0, 0, 0, 0, 0, 0, 0, 0};
  v8f accR = {0, 0, 0, 0, 0, 0, 0, 0};
#pragma unroll
  for (int s = 0; s < 6; ++s) {
    int kk = s >> 1;
    int ci0 = (s & 1) * 32;
    // B(k, n=t') = sA[(t'+kk)*64 + ci]  (contiguous k)
    v16bf bfrag = load_Bt_frag(sA + kk * 64, 64, ci0, 0);
    v16bf aP = load_A_frag(wT + (0 * 64 + w * 16) * 192, 192, s * 32);
    accP = wmma_bf16(aP, bfrag, accP);
    v16bf aQ = load_A_frag(wT + (1 * 64 + w * 16) * 192, 192, s * 32);
    accQ = wmma_bf16(aQ, bfrag, accQ);
    v16bf aR = load_A_frag(wT + (2 * 64 + w * 16) * 192, 192, s * 32);
    accR = wmma_bf16(aR, bfrag, accR);
  }
  int h = lane >> 4;
  int tp = lane & 15;           // t' within tile (N dim)
  int t = t0 + tp;
  int m0 = w * 16 + 8 * h;      // first output channel of this lane
  if (t < Tout) {
    float bp[8], bq[8], br[8];
#pragma unroll
    for (int v = 0; v < 8; ++v) {
      bp[v] = bias[m0 + v];
      bq[v] = bias[64 + m0 + v];
      br[v] = bias[128 + m0 + v];
    }
    v8bf pk;
#pragma unroll
    for (int v = 0; v < 8; ++v) {
      float P = accP[v] + bp[v], Q = accQ[v] + bq[v], R = accR[v] + br[v];
      pk[v] = (bf16_t)fmaxf(P * sigmoidf_(Q) + R, 0.f);
    }
    *(v8bf*)(out + (((size_t)b * Tout + t) * N + n) * 64 + m0) = pk;
  }
}

// ---------------------------------------------------------------------------
// ChebConv K=3 + bias + relu, all WMMA.
// Stage1: Tx1 = L*X        (A = L global, B = sXt channel-major)
// Stage2: Tx2 = 2*L*Tx1 -X (A = L global, B = sT1t channel-major)
// Stage3: Y^T = WcT * cat^T(A = WcT global, B = node-major sX/sT1/sT2)
// ---------------------------------------------------------------------------
__global__ __launch_bounds__(128) void cheb_kernel(
    const bf16_t* __restrict__ in, const bf16_t* __restrict__ Lg,
    const bf16_t* __restrict__ WcT, const float* __restrict__ bias,
    bf16_t* __restrict__ out, int B, int T) {
  int t = blockIdx.x, b = blockIdx.y;

  __shared__ __align__(16) bf16_t sX[64 * 64];    // node-major [node][c]
  __shared__ __align__(16) bf16_t sXt[64 * 64];   // channel-major [c][node]
  __shared__ __align__(16) bf16_t sT1t[64 * 64];  // [c][dst]
  __shared__ __align__(16) bf16_t sT1[48 * 64];   // [dst][c]
  __shared__ __align__(16) bf16_t sT2[48 * 64];   // [dst][c]

  int tid = threadIdx.x;
  const bf16_t* xin = in + (((size_t)b * T + t) * 33) * 64;
  const uint32* src32 = (const uint32*)xin;
  uint32* sX32 = (uint32*)sX;
  uint32* z1 = (uint32*)sXt;
  uint32* z2 = (uint32*)sT1t;
  for (int i = tid; i < 64 * 32; i += 128) {
    z1[i] = 0u;
    z2[i] = 0u;
    sX32[i] = (i < 33 * 32) ? src32[i] : 0u;
  }
  __syncthreads();
  // transpose X into sXt (one-time scattered)
  for (int i = tid; i < 33 * 64; i += 128) {
    int node = i >> 6, c = i & 63;
    sXt[c * 64 + node] = sX[i];
  }
  __syncthreads();

  int w = tid >> 5, lane = tid & 31;
  int h = lane >> 4;

  // ---- Stage 1: Tx1 = L*X  (m=dst, n=c, k=src) ----
#pragma unroll
  for (int it = 0; it < 3; ++it) {
    int tile = w * 3 + it;
    int mi = tile >> 2, ni = tile & 3;
    v8f acc = {0, 0, 0, 0, 0, 0, 0, 0};
#pragma unroll
    for (int ks = 0; ks < 2; ++ks) {
      v16bf a = load_A_frag(Lg + mi * 16 * 64, 64, ks * 32);
      v16bf bb = load_Bt_frag(sXt, 64, ks * 32, ni * 16);
      acc = wmma_bf16(a, bb, acc);
    }
    int nn = ni * 16 + (lane & 15);     // channel
    int m0 = mi * 16 + 8 * h;           // dst node base
    v8bf pk;
#pragma unroll
    for (int v = 0; v < 8; ++v) pk[v] = (bf16_t)acc[v];
    *(v8bf*)(sT1t + nn * 64 + m0) = pk;           // merged, [c][dst]
#pragma unroll
    for (int v = 0; v < 8; ++v) sT1[(m0 + v) * 64 + nn] = pk[v];  // [dst][c]
  }
  __syncthreads();

  // ---- Stage 2: Tx2 = 2*L*Tx1 - X ----
#pragma unroll
  for (int it = 0; it < 3; ++it) {
    int tile = w * 3 + it;
    int mi = tile >> 2, ni = tile & 3;
    v8f acc = {0, 0, 0, 0, 0, 0, 0, 0};
#pragma unroll
    for (int ks = 0; ks < 2; ++ks) {
      v16bf a = load_A_frag(Lg + mi * 16 * 64, 64, ks * 32);
      v16bf bb = load_Bt_frag(sT1t, 64, ks * 32, ni * 16);
      acc = wmma_bf16(a, bb, acc);
    }
    int nn = ni * 16 + (lane & 15);
    int m0 = mi * 16 + 8 * h;
    v8bf xv = *(const v8bf*)(sXt + nn * 64 + m0);  // X[m][nn], contiguous in m
#pragma unroll
    for (int v = 0; v < 8; ++v)
      sT2[(m0 + v) * 64 + nn] = (bf16_t)(2.0f * acc[v] - (float)xv[v]);
  }
  __syncthreads();

  // ---- Stage 3: Y^T[d][node] = sum_k WcT[d][k] * cat[node][k] ----
#pragma unroll
  for (int it = 0; it < 3; ++it) {
    int tile = w * 3 + it;
    int di = tile & 3, nodei = tile >> 2;
    v8f acc = {0, 0, 0, 0, 0, 0, 0, 0};
#pragma unroll
    for (int s = 0; s < 6; ++s) {
      int blk = s >> 1;
      const bf16_t* srcB = (blk == 0) ? sX : ((blk == 1) ? sT1 : sT2);
      v16bf a = load_A_frag(WcT + di * 16 * 192, 192, s * 32);
      v16bf bb = load_Bt_frag(srcB, 64, (s & 1) * 32, nodei * 16);
      acc = wmma_bf16(a, bb, acc);
    }
    int node = nodei * 16 + (lane & 15);
    if (node < 33) {
      int m0 = di * 16 + 8 * h;  // output channel base
      v8bf pk;
#pragma unroll
      for (int v = 0; v < 8; ++v)
        pk[v] = (bf16_t)fmaxf(acc[v] + bias[m0 + v], 0.f);
      *(v8bf*)(out + (((size_t)b * T + t) * 33 + node) * 64 + m0) = pk;
    }
  }
}

// ---------------------------------------------------------------------------
// BatchNorm per node: stats over (B,T,C), then apply + relu. Vectorized.
// ---------------------------------------------------------------------------
__global__ void bn_reduce_kernel(const bf16_t* __restrict__ in,
                                 float* __restrict__ stats, long rows) {
  int n = blockIdx.x;
  __shared__ float r1[256];
  __shared__ float r2[256];
  float a = 0.f, sq = 0.f;
  long groups = rows * 8;  // 8 channels per group
  for (long i = threadIdx.x; i < groups; i += 256) {
    long r = i >> 3;
    int g = (int)(i & 7);
    v8bf d = *(const v8bf*)(in + (r * 33 + n) * 64 + g * 8);
#pragma unroll
    for (int j = 0; j < 8; ++j) {
      float v = (float)d[j];
      a += v;
      sq += v * v;
    }
  }
  r1[threadIdx.x] = a;
  r2[threadIdx.x] = sq;
  __syncthreads();
  for (int st = 128; st > 0; st >>= 1) {
    if (threadIdx.x < st) {
      r1[threadIdx.x] += r1[threadIdx.x + st];
      r2[threadIdx.x] += r2[threadIdx.x + st];
    }
    __syncthreads();
  }
  if (threadIdx.x == 0) {
    float total = (float)(rows * 64);
    float mean = r1[0] / total;
    float var = r2[0] / total - mean * mean;
    stats[n] = mean;
    stats[33 + n] = rsqrtf(var + 1e-5f);
  }
}

__global__ void bn_apply_relu_kernel(const bf16_t* __restrict__ in,
                                     const float* __restrict__ stats,
                                     const float* __restrict__ g,
                                     const float* __restrict__ bta,
                                     bf16_t* __restrict__ out, long groups) {
  long i = (long)blockIdx.x * blockDim.x + threadIdx.x;
  if (i >= groups) return;
  long e0 = i * 8;
  int n = (int)((e0 >> 6) % 33);
  float mu = stats[n], rs = stats[33 + n], gg = g[n], bb = bta[n];
  v8bf d = *(const v8bf*)(in + e0);
  v8bf o;
#pragma unroll
  for (int j = 0; j < 8; ++j)
    o[j] = (bf16_t)fmaxf(((float)d[j] - mu) * rs * gg + bb, 0.f);
  *(v8bf*)(out + e0) = o;
}

// ---------------------------------------------------------------------------
// Temporal mean pooling -> (B, 33*64) bf16 ; 8 channels per thread
// ---------------------------------------------------------------------------
__global__ void tmean_kernel(const bf16_t* __restrict__ in,
                             bf16_t* __restrict__ pooled, int B, int T) {
  int i = blockIdx.x * blockDim.x + threadIdx.x;
  if (i >= B * 33 * 8) return;
  int g = i & 7, n = (i >> 3) % 33, b = i / (33 * 8);
  float s[8];
#pragma unroll
  for (int j = 0; j < 8; ++j) s[j] = 0.f;
  for (int t = 0; t < T; ++t) {
    v8bf d = *(const v8bf*)(in + (((size_t)b * T + t) * 33 + n) * 64 + g * 8);
#pragma unroll
    for (int j = 0; j < 8; ++j) s[j] += (float)d[j];
  }
  v8bf o;
  float inv = 1.0f / (float)T;
#pragma unroll
  for (int j = 0; j < 8; ++j) o[j] = (bf16_t)(s[j] * inv);
  *(v8bf*)(pooled + b * 2112 + n * 64 + g * 8) = o;
}

// ---------------------------------------------------------------------------
// FC1 (32x2112x256) WMMA + bias + relu. A,pooled and B,fc1wT from global.
// ---------------------------------------------------------------------------
__global__ __launch_bounds__(32) void fc1_kernel(const bf16_t* __restrict__ A,
                                                 const bf16_t* __restrict__ BwT,
                                                 const float* __restrict__ bias,
                                                 float* __restrict__ out) {
  int mi = blockIdx.x >> 4, ni = blockIdx.x & 15;
  v8f acc = {0, 0, 0, 0, 0, 0, 0, 0};
  for (int s = 0; s < 66; ++s) {
    v16bf a = load_A_frag(A + mi * 16 * 2112, 2112, s * 32);
    v16bf b = load_Bt_frag(BwT, 2112, s * 32, ni * 16);
    acc = wmma_bf16(a, b, acc);
  }
  int lane = threadIdx.x & 31;
  int c = ni * 16 + (lane & 15);
  float bb = bias[c];
#pragma unroll
  for (int v = 0; v < 8; ++v) {
    int m = mi * 16 + ((lane >> 4) << 3) + v;
    out[m * 256 + c] = fmaxf(acc[v] + bb, 0.f);
  }
}

__global__ void fc2_kernel(const float* __restrict__ h,
                           const float* __restrict__ w,
                           const float* __restrict__ b,
                           float* __restrict__ out) {
  int i = blockIdx.x * blockDim.x + threadIdx.x;
  if (i >= 320) return;
  int bb = i / 10, o = i % 10;
  float s = b[o];
  for (int k = 0; k < 256; ++k) s += h[bb * 256 + k] * w[k * 10 + o];
  out[i] = s;
}

// ---------------------------------------------------------------------------
// Launch
// ---------------------------------------------------------------------------
extern "C" void kernel_launch(void* const* d_in, const int* in_sizes, int n_in,
                              void* d_out, int out_size, void* d_ws, size_t ws_size,
                              hipStream_t stream) {
  (void)in_sizes; (void)n_in; (void)out_size; (void)ws_size;
  const float* x      = (const float*)d_in[0];
  const int*   edge   = (const int*)d_in[1];
  const float* s1tc1w = (const float*)d_in[2];
  const float* s1tc1b = (const float*)d_in[3];
  const float* s1chw  = (const float*)d_in[4];
  const float* s1chb  = (const float*)d_in[5];
  const float* s1tc2w = (const float*)d_in[6];
  const float* s1tc2b = (const float*)d_in[7];
  const float* s1bng  = (const float*)d_in[8];
  const float* s1bnb  = (const float*)d_in[9];
  const float* s2tc1w = (const float*)d_in[10];
  const float* s2tc1b = (const float*)d_in[11];
  const float* s2chw  = (const float*)d_in[12];
  const float* s2chb  = (const float*)d_in[13];
  const float* s2tc2w = (const float*)d_in[14];
  const float* s2tc2b = (const float*)d_in[15];
  const float* s2bng  = (const float*)d_in[16];
  const float* s2bnb  = (const float*)d_in[17];
  const float* fc1w   = (const float*)d_in[18];
  const float* fc1b   = (const float*)d_in[19];
  const float* fc2w   = (const float*)d_in[20];
  const float* fc2b   = (const float*)d_in[21];

  char* ws = (char*)d_ws;
  const size_t OFF_B = 69206016ull;  // per activation buffer
  bf16_t* bufA = (bf16_t*)ws;
  bf16_t* bufB = (bf16_t*)(ws + OFF_B);
  char* sm = ws + 2 * OFF_B;
  bf16_t* Lb     = (bf16_t*)(sm);            // 8192 B
  bf16_t* wtcT   = (bf16_t*)(sm + 8192);     // 3 * 73728 B
  bf16_t* wcatT1 = (bf16_t*)(sm + 229376);   // 24576 B
  bf16_t* wcatT2 = (bf16_t*)(sm + 253952);   // 24576 B
  bf16_t* fc1wT  = (bf16_t*)(sm + 278528);   // 1081344 B
  float*  stats  = (float*)(sm + 1359872);   // 264 B
  bf16_t* pooled = (bf16_t*)(sm + 1360896);  // 135168 B
  float*  fc1out = (float*)(sm + 1497088);   // 32768 B
  float*  outp   = (float*)d_out;

  const int B = 32;

  prep_kernel<<<128, 256, 0, stream>>>(edge, s1tc2w, s2tc1w, s2tc2w,
                                       s1chw, s2chw, fc1w,
                                       Lb, wtcT, wcatT1, wcatT2, fc1wT);

  // ---- STConv block 1 ----
  tconv3_kernel<<<dim3(B * 510, 33), 64, 0, stream>>>(x, s1tc1w, s1tc1b, bufA, B, 512);
  cheb_kernel<<<dim3(510, B), 128, 0, stream>>>(bufA, Lb, wcatT1, s1chb, bufB, B, 510);
  tconv_gemm_kernel<<<dim3(32, 33, B), 128, 0, stream>>>(bufB, wtcT, s1tc2b, bufA, B, 510);
  bn_reduce_kernel<<<33, 256, 0, stream>>>(bufA, stats, (long)B * 508);
  {
    long groups = (long)B * 508 * 33 * 8;
    bn_apply_relu_kernel<<<(unsigned)((groups + 255) / 256), 256, 0, stream>>>(
        bufA, stats, s1bng, s1bnb, bufB, groups);
  }

  // ---- STConv block 2 ----
  tconv_gemm_kernel<<<dim3(32, 33, B), 128, 0, stream>>>(bufB, wtcT + 36864, s2tc1b, bufA, B, 508);
  cheb_kernel<<<dim3(506, B), 128, 0, stream>>>(bufA, Lb, wcatT2, s2chb, bufB, B, 506);
  tconv_gemm_kernel<<<dim3(32, 33, B), 128, 0, stream>>>(bufB, wtcT + 2 * 36864, s2tc2b, bufA, B, 506);
  bn_reduce_kernel<<<33, 256, 0, stream>>>(bufA, stats, (long)B * 504);
  {
    long groups = (long)B * 504 * 33 * 8;
    bn_apply_relu_kernel<<<(unsigned)((groups + 255) / 256), 256, 0, stream>>>(
        bufA, stats, s2bng, s2bnb, bufB, groups);
  }

  // ---- Head ----
  tmean_kernel<<<(B * 33 * 8 + 255) / 256, 256, 0, stream>>>(bufB, pooled, B, 504);
  fc1_kernel<<<32, 32, 0, stream>>>(pooled, fc1wT, fc1b, fc1out);
  fc2_kernel<<<5, 64, 0, stream>>>(fc1out, fc2w, fc2b, outp);
}
